// PostProcessor_56238301773971
// MI455X (gfx1250) — compile-verified, matrix operand order checked by reading, not run on hardware
//
#include <hip/hip_runtime.h>
#include <hip/hip_bf16.h>

// ---------------- problem constants (match reference) ----------------
#define B_IMG   16
#define R_PROP  4096
#define C_CLS   81
#define NCLS    80            // C-1 (background dropped)
#define K_CAND  300
#define K_PAD   320           // padded to 20 tiles of 16 for WMMA
#define D_DET   100
#define FLAT_N  (NCLS * K_CAND)   // 24000 per image
#define W_IMG_M1 1332.0f
#define H_IMG_M1 799.0f
#define BBOX_CLIP 4.135166556742356f   // log(1000/16)
#define SCORE_TH 0.05f

typedef __attribute__((ext_vector_type(16))) _Float16 v16h;
typedef __attribute__((ext_vector_type(8)))  float    v8f;

// ---------------------------------------------------------------------
// Kernel 1: per-row softmax stats (rowmax, 1/sum(exp)) so we never
// materialize the [B*R, 81] probability tensor.
// ---------------------------------------------------------------------
__global__ void row_stats_kernel(const float* __restrict__ logits,
                                 float* __restrict__ rowmax,
                                 float* __restrict__ invsum) {
  int row = blockIdx.x * blockDim.x + threadIdx.x;
  if (row >= B_IMG * R_PROP) return;
  const float* lp = logits + (long)row * C_CLS;
  __builtin_prefetch(lp, 0, 0);                 // global_prefetch_b8
  float m = -1e30f;
  for (int c = 0; c < C_CLS; ++c) m = fmaxf(m, lp[c]);
  float s = 0.f;
  for (int c = 0; c < C_CLS; ++c) s += __expf(lp[c] - m);
  rowmax[row] = m;
  invsum[row] = 1.0f / s;
}

// ---------------------------------------------------------------------
// Kernel 2: one workgroup per (image, class). Full bitonic sort of the
// 4096 (score, idx) pairs in LDS (32 KB of the 320 KB/WGP budget),
// descending, ties -> lower index (matches jax top_k). Decode + clip
// boxes only for the 300 winners.
// ---------------------------------------------------------------------
__global__ void topk_decode_kernel(const float* __restrict__ logits,
                                   const float* __restrict__ boxreg,
                                   const float* __restrict__ props,
                                   const float* __restrict__ rowmax,
                                   const float* __restrict__ invsum,
                                   float* __restrict__ top_s,
                                   float* __restrict__ top_b) {
  __shared__ float skey[R_PROP];
  __shared__ int   sidx[R_PROP];
  const int b   = blockIdx.x / NCLS;
  const int cc  = blockIdx.x % NCLS;
  const int c   = cc + 1;
  const int tid = threadIdx.x;

  for (int r = tid; r < R_PROP; r += blockDim.x) {
    long row = (long)b * R_PROP + r;
    float l = logits[row * C_CLS + c];
    skey[r] = __expf(l - rowmax[row]) * invsum[row];
    sidx[r] = r;
  }
  __syncthreads();

  for (int k = 2; k <= R_PROP; k <<= 1) {
    for (int j = k >> 1; j >= 1; j >>= 1) {
      for (int i = tid; i < R_PROP; i += blockDim.x) {
        int p = i ^ j;
        if (p > i) {
          float si = skey[i], sp = skey[p];
          int   ii = sidx[i], ip = sidx[p];
          bool iFirst = (si > sp) || (si == sp && ii < ip); // desired desc order
          bool desc   = ((i & k) == 0);
          if (desc != iFirst) {
            skey[i] = sp; skey[p] = si;
            sidx[i] = ip; sidx[p] = ii;
          }
        }
      }
      __syncthreads();
    }
  }

  for (int t = tid; t < K_CAND; t += blockDim.x) {
    int r = sidx[t];
    float sc = skey[t];
    long row = (long)b * R_PROP + r;
    float x1 = props[row * 4 + 0], y1 = props[row * 4 + 1];
    float x2 = props[row * 4 + 2], y2 = props[row * 4 + 3];
    float w = x2 - x1 + 1.f, h = y2 - y1 + 1.f;
    float cx = x1 + 0.5f * w, cy = y1 + 0.5f * h;
    const float* rp = boxreg + row * (C_CLS * 4) + c * 4;
    float dx = rp[0] * 0.1f;
    float dy = rp[1] * 0.1f;
    float dw = fminf(rp[2] * 0.2f, BBOX_CLIP);
    float dh = fminf(rp[3] * 0.2f, BBOX_CLIP);
    float pcx = dx * w + cx, pcy = dy * h + cy;
    float pw = __expf(dw) * w, ph = __expf(dh) * h;
    float nx1 = pcx - 0.5f * pw, ny1 = pcy - 0.5f * ph;
    float nx2 = pcx + 0.5f * pw - 1.f, ny2 = pcy + 0.5f * ph - 1.f;
    nx1 = fminf(fmaxf(nx1, 0.f), W_IMG_M1);
    ny1 = fminf(fmaxf(ny1, 0.f), H_IMG_M1);
    nx2 = fminf(fmaxf(nx2, 0.f), W_IMG_M1);
    ny2 = fminf(fmaxf(ny2, 0.f), H_IMG_M1);
    long o = (long)blockIdx.x * K_CAND + t;
    top_s[o] = sc;
    top_b[o * 4 + 0] = nx1; top_b[o * 4 + 1] = ny1;
    top_b[o * 4 + 2] = nx2; top_b[o * 4 + 3] = ny2;
  }
}

// ---------------------------------------------------------------------
// Kernel 3: NMS per (image, class). Build the full 320x320 (IoU>0.5)
// bitmask in parallel; the pairwise (area_i + area_j) term for every
// 16x16 tile is produced by ONE v_wmma_f32_16x16x32_f16 (areas hi/lo
// split into f16, scaled by 1/64; rank-4 product => a_i + a_j in the
// f32 accumulator). The 8 wave32 ballots of a tile are wave-uniform, so
// bit packing is branch-free: lane l<16 owns row ti*16+l and stores its
// 16-bit segment (ballot[l&7], low/high half) as one u16 -- each tile
// owns mask column tj exclusively, so no atomics, one divergence region.
// Then a 300-step serial 10-word bit sweep replicates greedy NMS.
// ---------------------------------------------------------------------
__global__ void nms_kernel(const float* __restrict__ top_s,
                           const float* __restrict__ top_b,
                           float* __restrict__ flat_s) {
  __shared__ float4 sbox[K_PAD];
  __shared__ float  ss[K_PAD];
  __shared__ _Float16 ahi[K_PAD], alo[K_PAD];
  __shared__ unsigned short mask16[K_PAD * 20];   // 320 rows x 20 u16 segments
  __shared__ unsigned char keepf[K_PAD];
  const int tid = threadIdx.x;
  const long base = (long)blockIdx.x * K_CAND;

  for (int t = tid; t < K_PAD; t += blockDim.x) {
    float x1 = 0.f, y1 = 0.f, x2 = 0.f, y2 = 0.f, sc = -1e30f;
    if (t < K_CAND) {
      x1 = top_b[(base + t) * 4 + 0];
      y1 = top_b[(base + t) * 4 + 1];
      x2 = top_b[(base + t) * 4 + 2];
      y2 = top_b[(base + t) * 4 + 3];
      sc = top_s[base + t];
    }
    sbox[t] = make_float4(x1, y1, x2, y2);
    ss[t] = sc;
    float area = (x2 - x1 + 1.f) * (y2 - y1 + 1.f) * 0.015625f;  // /64 fits f16
    _Float16 hi = (_Float16)area;
    ahi[t] = hi;
    alo[t] = (_Float16)(area - (float)hi);
  }
  for (int i = tid; i < K_PAD * 20; i += blockDim.x) mask16[i] = 0;
  __syncthreads();

  const int lane = tid & 31;
  const int wave = tid >> 5;
  const int N = lane & 15;
  // 400 tiles / 8 waves = 50 each: fully uniform EXEC for WMMA + ballots.
  for (int t = wave; t < 400; t += (int)(blockDim.x >> 5)) {
    int ti = t / 20, tj = t % 20;
    v16h a  = {};
    v16h bm = {};
    if (lane < 16) {
      int ri = ti * 16 + lane;           // A row M=lane, K=0..3
      a[0] = ahi[ri]; a[1] = alo[ri];
      a[2] = (_Float16)1.0f; a[3] = (_Float16)1.0f;
      int rj = tj * 16 + lane;           // B col N=lane, K=0..3
      bm[0] = (_Float16)1.0f; bm[1] = (_Float16)1.0f;
      bm[2] = ahi[rj]; bm[3] = alo[rj];
    }
    v8f cacc = {};
    cacc = __builtin_amdgcn_wmma_f32_16x16x32_f16(
        false, a, false, bm, (short)0, cacc, false, false);

    // column box for this lane (invariant over k)
    const int j = tj * 16 + N;
    const float4 cb = sbox[j];

    unsigned int bal[8];
#pragma unroll
    for (int k = 0; k < 8; ++k) {
      int M = (lane < 16) ? k : (k + 8);   // C/D layout: VGPR k, lane halves
      int i = ti * 16 + M;
      float4 rb = sbox[i];                 // ds_load_b128 (broadcast per half)
      float iw = fminf(rb.z, cb.z) - fmaxf(rb.x, cb.x) + 1.f;
      float ih = fminf(rb.w, cb.w) - fmaxf(rb.y, cb.y) + 1.f;
      float inter = fmaxf(iw, 0.f) * fmaxf(ih, 0.f);
      float uni = cacc[k] * 64.f - inter;          // area_i + area_j - inter
      bal[k] = __builtin_amdgcn_ballot_w32(inter > 0.5f * uni);
    }

    // branch-free selection: lane l<16 -> row ti*16+l, segment from bal[l&7]
    const int kk = lane & 7;
    unsigned int sel = bal[0];
#pragma unroll
    for (int q = 1; q < 8; ++q) sel = (kk == q) ? bal[q] : sel;
    unsigned int seg = (lane < 8) ? (sel & 0xFFFFu) : (sel >> 16);
    if (lane < 16)
      mask16[(ti * 16 + lane) * 20 + tj] = (unsigned short)seg;
  }
  __syncthreads();

  if (tid == 0) {  // 300 x 10-word greedy bit sweep
    unsigned int sp[10];
#pragma unroll
    for (int w = 0; w < 10; ++w) sp[w] = 0u;
    for (int i = 0; i < K_CAND; ++i) {
      bool kept = (ss[i] > SCORE_TH) && !((sp[i >> 5] >> (i & 31)) & 1u);
      keepf[i] = kept ? 1 : 0;
      if (kept) {
        const unsigned int* mrow = (const unsigned int*)&mask16[i * 20];
#pragma unroll
        for (int w = 0; w < 10; ++w) sp[w] |= mrow[w];
      }
    }
  }
  __syncthreads();

  for (int t = tid; t < K_CAND; t += blockDim.x)
    flat_s[base + t] = keepf[t] ? ss[t] : -1.0f;
}

// ---------------------------------------------------------------------
// Kernel 4: per-image top-100 of 24000 (L2-hot). 100 argmax passes with
// jax-top_k tie-breaking (lower index wins). Taken-set lives in LDS so
// no cross-replay global state is left behind.
// ---------------------------------------------------------------------
__global__ void final_topk_kernel(const float* __restrict__ flat_s,
                                  const float* __restrict__ top_b,
                                  float* __restrict__ out_boxes,
                                  float* __restrict__ out_scores,
                                  int* __restrict__ out_labels) {
  __shared__ unsigned int taken[(FLAT_N + 31) / 32];
  __shared__ float rs[256];
  __shared__ int   ri[256];
  const int b = blockIdx.x;
  const int tid = threadIdx.x;
  for (int i = tid; i < (FLAT_N + 31) / 32; i += blockDim.x) taken[i] = 0u;
  __syncthreads();
  const float* fs = flat_s + (long)b * FLAT_N;
  const float* fb = top_b + (long)b * FLAT_N * 4;

  for (int d = 0; d < D_DET; ++d) {
    float bs = -1e30f; int bi = -1;
    for (int i = tid; i < FLAT_N; i += blockDim.x) {
      if ((taken[i >> 5] >> (i & 31)) & 1u) continue;
      float s = fs[i];
      if (s > bs || (s == bs && (unsigned)i < (unsigned)bi)) { bs = s; bi = i; }
    }
    rs[tid] = bs; ri[tid] = bi;
    __syncthreads();
    for (int off = 128; off >= 1; off >>= 1) {
      if (tid < off) {
        float s2 = rs[tid + off]; int i2 = ri[tid + off];
        if (i2 != -1 &&
            (s2 > rs[tid] || (s2 == rs[tid] && (ri[tid] == -1 || i2 < ri[tid])))) {
          rs[tid] = s2; ri[tid] = i2;
        }
      }
      __syncthreads();
    }
    int best = ri[0]; float bscore = rs[0];
    if (tid == 0) {
      taken[best >> 5] |= (1u << (best & 31));
      out_scores[b * D_DET + d] = bscore;
      out_labels[b * D_DET + d] = (bscore > 0.f) ? (best / K_CAND + 1) : 0;
      long o = ((long)b * D_DET + d) * 4;
      long src = (long)best * 4;
      out_boxes[o + 0] = fb[src + 0];
      out_boxes[o + 1] = fb[src + 1];
      out_boxes[o + 2] = fb[src + 2];
      out_boxes[o + 3] = fb[src + 3];
    }
    __syncthreads();
  }
}

// ---------------------------------------------------------------------
extern "C" void kernel_launch(void* const* d_in, const int* in_sizes, int n_in,
                              void* d_out, int out_size, void* d_ws, size_t ws_size,
                              hipStream_t stream) {
  const float* logits = (const float*)d_in[0];   // [B*R, 81]
  const float* boxreg = (const float*)d_in[1];   // [B*R, 324]
  const float* props  = (const float*)d_in[2];   // [B*R, 4]

  // workspace layout (floats)
  float* ws      = (float*)d_ws;
  float* rowmax  = ws;                              // 65536
  float* invsum  = rowmax + B_IMG * R_PROP;         // 65536
  float* top_s   = invsum + B_IMG * R_PROP;         // 1280*300
  float* top_b   = top_s + (long)B_IMG * NCLS * K_CAND;        // 1280*300*4
  float* flat_s  = top_b + (long)B_IMG * NCLS * K_CAND * 4;    // 1280*300

  // outputs: boxes [16,100,4] f32 | scores [16,100] f32 | labels [16,100] i32
  float* out_boxes  = (float*)d_out;
  float* out_scores = out_boxes + B_IMG * D_DET * 4;
  int*   out_labels = (int*)(out_boxes + B_IMG * D_DET * 5);

  hipLaunchKernelGGL(row_stats_kernel, dim3((B_IMG * R_PROP) / 256), dim3(256),
                     0, stream, logits, rowmax, invsum);
  hipLaunchKernelGGL(topk_decode_kernel, dim3(B_IMG * NCLS), dim3(256),
                     0, stream, logits, boxreg, props, rowmax, invsum, top_s, top_b);
  hipLaunchKernelGGL(nms_kernel, dim3(B_IMG * NCLS), dim3(256),
                     0, stream, top_s, top_b, flat_s);
  hipLaunchKernelGGL(final_topk_kernel, dim3(B_IMG), dim3(256),
                     0, stream, flat_s, top_b, out_boxes, out_scores, out_labels);
}
